// Z100_MLA_41334765257148
// MI455X (gfx1250) — matrix operand order repaired and checked by reading, not compiled, not run on hardware
//
#include <hip/hip_runtime.h>
#include <math.h>

// ---------------------------------------------------------------------------
// MLA forward for MI455X (gfx1250): bf16 WMMA GEMMs (async-LDS double-buffered
// B operand) + flash attention.
// ---------------------------------------------------------------------------

#define S_LEN 2048
#define DIM   2048
#define NH    16
#define DQK   192      // 128 nope + 64 rope
#define DV    128
#define QLORA 1536
#define KVLORA 512

typedef __attribute__((ext_vector_type(16))) __bf16 v16bf;
typedef __attribute__((ext_vector_type(8)))  float  v8f;

union Frag { v16bf v; uint4 u[2]; };

__device__ __forceinline__ unsigned short f2bf(float f) {
  unsigned int u = __float_as_uint(f);
  u += 0x7FFFu + ((u >> 16) & 1u);          // round-to-nearest-even
  return (unsigned short)(u >> 16);
}

// -------------------------- f32 -> bf16 convert ----------------------------
__global__ void mla_cvt_bf16(const float* __restrict__ in,
                             unsigned short* __restrict__ out, int n) {
  int i = blockIdx.x * 256 + threadIdx.x;
  if (i < n) out[i] = f2bf(in[i]);
}

// ------------------- GEMM: C[MxN] = A[MxK] * B[NxK]^T ----------------------
// A,B bf16 row-major (K contiguous); C f32 row-major.
// 256 threads = 8 waves. Each wave computes a 16x64 strip (4 accumulators,
// one A-frag reused 4x). Waves arranged (8/WN) down M x WN across N:
//   WN=2: block tile 64x128 ; WN=1: block tile 128x64 (for N=576).
// B block tile (BN x 32 K-slice) is staged into LDS with
// global_load_async_to_lds_b128 (ASYNCcnt), double buffered: while slice s is
// multiplied, slice s+1 streams in. 2 async instr/wave/stage and async loads
// retire in order, so s_wait_asynccnt 2 releases the current buffer.
template <int WN>
__global__ __launch_bounds__(256) void mla_gemm_nt(
    const unsigned short* __restrict__ A, const unsigned short* __restrict__ B,
    float* __restrict__ C, int M, int N, int K) {
  constexpr int WM = 8 / WN;       // waves down M
  constexpr int BN = WN * 64;      // block tile cols
  __shared__ __align__(16) unsigned short Blds[2][BN][32];

  int tid  = threadIdx.x;
  int wave = tid >> 5;
  int lane = tid & 31;
  int wm = wave / WN, wn = wave % WN;
  int r  = lane & 15;
  int hi = lane >> 4;

  int tmb = blockIdx.y * (WM * 16);
  int tnb = blockIdx.x * BN;

  // A-frag source: lane = row; elems 0-15 = k0..k0+15 contiguous
  const unsigned short* arow = A + (size_t)(tmb + wm * 16 + r) * K + hi * 16;

  // Stage one BN x 32 slice of B into Blds[buf] (64 B per column, 32 B/thread,
  // first BN*2 threads participate -- wave aligned for both WN values).
  auto stage = [&](int buf, int k) {
    if (tid < BN * 2) {
      int col  = tid >> 1;
      int part = tid & 1;                                 // 0: +0B, 1: +32B
      const unsigned short* gp = B + (size_t)(tnb + col) * K + k + part * 16;
      unsigned ldsOff = (unsigned)(uintptr_t)&Blds[buf][col][part * 16];
      asm volatile("global_load_async_to_lds_b128 %0, %1, off"
                   :: "v"(ldsOff), "v"(gp) : "memory");
      asm volatile("global_load_async_to_lds_b128 %0, %1, off"
                   :: "v"(ldsOff + 16u), "v"(gp + 8) : "memory");
    }
  };

  v8f acc[4] = {};
  int nsteps = K / 32;
  stage(0, 0);
  for (int s = 0; s < nsteps; s++) {
    int k = s * 32;
    if (s + 1 < nsteps) {
      stage((s + 1) & 1, k + 32);
      asm volatile("s_wait_asynccnt 2" ::: "memory");   // current slice landed
    } else {
      asm volatile("s_wait_asynccnt 0" ::: "memory");
    }
    __syncthreads();                                    // all waves' data in LDS

    Frag a;
    a.u[0] = *(const uint4*)(arow + k);
    a.u[1] = *(const uint4*)(arow + k + 8);
    const unsigned short(*Bb)[32] = Blds[s & 1];
#pragma unroll
    for (int g = 0; g < 4; g++) {
      Frag b;   // B-frag: lane = col; elems 0-15 = K kb..kb+15 contiguous
      const unsigned short* bp = &Bb[wn * 64 + g * 16 + r][hi * 16];
      b.u[0] = *(const uint4*)bp;
      b.u[1] = *(const uint4*)(bp + 8);
      acc[g] = __builtin_amdgcn_wmma_f32_16x16x32_bf16(false, a.v, false, b.v,
                                                       (short)0, acc[g],
                                                       false, false);
    }
    __syncthreads();   // done reading this buffer before it is restaged
  }

  // D layout: element i -> row (i + 8*hi), col r
  float* crow = C + (size_t)(tmb + wm * 16 + hi * 8) * N + tnb + wn * 64 + r;
#pragma unroll
  for (int g = 0; g < 4; g++)
#pragma unroll
    for (int i = 0; i < 8; i++) crow[(size_t)i * N + g * 16] = acc[g][i];
}

// ------------------------- RMSNorm (row-wise) ------------------------------
__global__ void mla_rmsnorm(const float* __restrict__ in,
                            const float* __restrict__ w,
                            unsigned short* __restrict__ out, int width) {
  int row = blockIdx.x;
  const float* x = in + (size_t)row * width;
  float ss = 0.f;
  for (int i = threadIdx.x; i < width; i += 256) { float v = x[i]; ss += v * v; }
  __shared__ float red[256];
  red[threadIdx.x] = ss; __syncthreads();
  for (int s = 128; s > 0; s >>= 1) {
    if (threadIdx.x < s) red[threadIdx.x] += red[threadIdx.x + s];
    __syncthreads();
  }
  float scale = rsqrtf(red[0] / (float)width + 1e-6f);
  for (int i = threadIdx.x; i < width; i += 256)
    out[(size_t)row * width + i] = f2bf(x[i] * scale * w[i]);
}

// ------------- kv row: RMSNorm first 512, RoPE last 64 (pre-norm) ----------
__global__ void mla_kv_norm_rope(const float* __restrict__ kv,
                                 const float* __restrict__ w,
                                 const float* __restrict__ fcos,
                                 const float* __restrict__ fsin,
                                 unsigned short* __restrict__ kvc,
                                 float* __restrict__ kpe) {
  int s = blockIdx.x;
  const float* x = kv + (size_t)s * 576;
  float ss = 0.f;
  for (int i = threadIdx.x; i < KVLORA; i += 256) { float v = x[i]; ss += v * v; }
  __shared__ float red[256];
  red[threadIdx.x] = ss; __syncthreads();
  for (int t = 128; t > 0; t >>= 1) {
    if (threadIdx.x < t) red[threadIdx.x] += red[threadIdx.x + t];
    __syncthreads();
  }
  float scale = rsqrtf(red[0] / (float)KVLORA + 1e-6f);
  for (int i = threadIdx.x; i < KVLORA; i += 256)
    kvc[(size_t)s * KVLORA + i] = f2bf(x[i] * scale * w[i]);
  if (threadIdx.x < 32) {
    int j = threadIdx.x;
    float c = fcos[s * 32 + j], sn = fsin[s * 32 + j];
    float x1 = x[KVLORA + 2 * j], x2 = x[KVLORA + 2 * j + 1];
    kpe[(size_t)s * 64 + 2 * j]     = x1 * c - x2 * sn;
    kpe[(size_t)s * 64 + 2 * j + 1] = x1 * sn + x2 * c;
  }
}

// --------- build q: RoPE the pe part, fold softmax scale, -> bf16 ----------
// qf: (S, H*192) f32 ; qk: (S, H, 192) bf16
__global__ void mla_build_q(const float* __restrict__ qf,
                            const float* __restrict__ fcos,
                            const float* __restrict__ fsin,
                            unsigned short* __restrict__ qk) {
  int s = blockIdx.x;
  const float scale = 0.07216878364870322f;     // 1/sqrt(192)
  const float* q = qf + (size_t)s * (NH * DQK);
  unsigned short* o = qk + (size_t)s * (NH * DQK);
  for (int p = threadIdx.x; p < NH * 96; p += 256) {
    int h = p / 96, idx = p % 96, base = h * DQK;
    if (idx < 64) {
      int d = idx * 2;
      o[base + d]     = f2bf(q[base + d] * scale);
      o[base + d + 1] = f2bf(q[base + d + 1] * scale);
    } else {
      int j = idx - 64;
      float c = fcos[s * 32 + j], sn = fsin[s * 32 + j];
      float x1 = q[base + 128 + 2 * j], x2 = q[base + 128 + 2 * j + 1];
      o[base + 128 + 2 * j]     = f2bf((x1 * c - x2 * sn) * scale);
      o[base + 128 + 2 * j + 1] = f2bf((x1 * sn + x2 * c) * scale);
    }
  }
}

// ------- build k (nope || broadcast pe) bf16 and v^T (H,128,S) bf16 --------
// kvb: (S, H, 256) f32 : [0:128] k_nope, [128:256] v
__global__ void mla_build_k(const float* __restrict__ kvb,
                            const float* __restrict__ kpe,
                            unsigned short* __restrict__ kbf,
                            unsigned short* __restrict__ vt) {
  int s = blockIdx.x;
  const float* r = kvb + (size_t)s * (NH * 256);
  for (int idx = threadIdx.x; idx < NH * 256; idx += 256) {
    int h = idx >> 8, d2 = idx & 255;
    float v = r[h * 256 + d2];
    if (d2 < 128)
      kbf[((size_t)s * NH + h) * DQK + d2] = f2bf(v);
    else
      vt[((size_t)(h * DV + (d2 - 128))) * S_LEN + s] = f2bf(v);
  }
  for (int idx = threadIdx.x; idx < NH * 64; idx += 256) {
    int h = idx >> 6, e = idx & 63;
    kbf[((size_t)s * NH + h) * DQK + 128 + e] = f2bf(kpe[(size_t)s * 64 + e]);
  }
}

// ---------------------- flash attention, causal ----------------------------
// One wave (32 threads) per (query-tile of 16, head). Keys in chunks of 32:
// 2 score tiles (6 WMMA each over K=192) -> online softmax -> P staged in LDS
// (D-layout -> A-layout swizzle) -> 8 PV WMMAs against v^T.
__global__ __launch_bounds__(32) void mla_attn(
    const unsigned short* __restrict__ qk,   // (S,H,192) bf16, scale folded
    const unsigned short* __restrict__ kbf,  // (S,H,192) bf16
    const unsigned short* __restrict__ vt,   // (H,128,S) bf16
    unsigned short* __restrict__ attn) {     // (S, H*128) bf16
  int qt = blockIdx.x;          // query tile (16 rows)
  int h  = blockIdx.y;
  int lane = threadIdx.x;
  int nIdx = lane & 15;
  int hi   = lane >> 4;
  int rowOff = hi * 8;

  __shared__ __align__(16) unsigned short Pp[16][32];

  // Q A-fragments for the whole 16x192 tile (6 K-steps of 32)
  Frag qf[6];
  {
    const unsigned short* qp =
        qk + ((size_t)(qt * 16 + nIdx) * NH + h) * DQK;
#pragma unroll
    for (int c = 0; c < 6; c++) {
      int ka = c * 32 + hi * 16;
      qf[c].u[0] = *(const uint4*)(qp + ka);
      qf[c].u[1] = *(const uint4*)(qp + ka + 8);
    }
  }

  v8f accO[8] = {};           // 16x128 output, 8 column groups
  float Mr[8], Lr[8];
#pragma unroll
  for (int i = 0; i < 8; i++) { Mr[i] = -1e30f; Lr[i] = 0.f; }

  int lastKey = qt * 16 + 15;
  for (int kk = 0; kk <= lastKey; kk += 32) {
    // ---- two 16x16 score tiles over full d=192 ----
    v8f s0 = {}, s1 = {};
    {
      const unsigned short* kp0 =
          kbf + ((size_t)(kk + nIdx) * NH + h) * DQK + hi * 16;
      const unsigned short* kp1 =
          kbf + ((size_t)(kk + 16 + nIdx) * NH + h) * DQK + hi * 16;
#pragma unroll
      for (int c = 0; c < 6; c++) {
        Frag b;
        b.u[0] = *(const uint4*)(kp0 + c * 32);
        b.u[1] = *(const uint4*)(kp0 + c * 32 + 8);
        s0 = __builtin_amdgcn_wmma_f32_16x16x32_bf16(false, qf[c].v, false, b.v,
                                                     (short)0, s0, false, false);
      }
#pragma unroll
      for (int c = 0; c < 6; c++) {
        Frag b;
        b.u[0] = *(const uint4*)(kp1 + c * 32);
        b.u[1] = *(const uint4*)(kp1 + c * 32 + 8);
        s1 = __builtin_amdgcn_wmma_f32_16x16x32_bf16(false, qf[c].v, false, b.v,
                                                     (short)0, s1, false, false);
      }
    }
    // ---- causal mask + online softmax (rows = element index) ----
    int k0 = kk + nIdx, k1 = kk + 16 + nIdx;
#pragma unroll
    for (int i = 0; i < 8; i++) {
      int row = qt * 16 + rowOff + i;
      float a0 = (k0 <= row) ? s0[i] : -1e30f;
      float a1 = (k1 <= row) ? s1[i] : -1e30f;
      float cm = fmaxf(a0, a1);
#pragma unroll
      for (int m = 1; m < 16; m <<= 1) cm = fmaxf(cm, __shfl_xor(cm, m, 32));
      float nM = fmaxf(Mr[i], cm);
      float alpha = __expf(Mr[i] - nM);
      float p0 = __expf(a0 - nM);
      float p1 = __expf(a1 - nM);
      float ls = p0 + p1;
#pragma unroll
      for (int m = 1; m < 16; m <<= 1) ls += __shfl_xor(ls, m, 32);
      Lr[i] = Lr[i] * alpha + ls;
      Mr[i] = nM;
#pragma unroll
      for (int g = 0; g < 8; g++) accO[g][i] *= alpha;
      Pp[rowOff + i][nIdx]      = f2bf(p0);
      Pp[rowOff + i][16 + nIdx] = f2bf(p1);
    }
    __syncthreads();
    // ---- P (16x32) @ V (32x128): A-frag from LDS, B-frag from v^T ----
    Frag pf;
    {
      int pk = hi * 16;
      pf.u[0] = *(const uint4*)&Pp[nIdx][pk];
      pf.u[1] = *(const uint4*)&Pp[nIdx][pk + 8];
    }
#pragma unroll
    for (int g = 0; g < 8; g++) {
      Frag b;
      const unsigned short* vp =
          vt + ((size_t)(h * DV + g * 16 + nIdx)) * S_LEN + kk + hi * 16;
      b.u[0] = *(const uint4*)vp;
      b.u[1] = *(const uint4*)(vp + 8);
      accO[g] = __builtin_amdgcn_wmma_f32_16x16x32_bf16(false, pf.v, false, b.v,
                                                        (short)0, accO[g],
                                                        false, false);
    }
    __syncthreads();
  }

  // ---- epilogue: normalize and store bf16 ----
#pragma unroll
  for (int g = 0; g < 8; g++) {
#pragma unroll
    for (int i = 0; i < 8; i++) {
      int row = qt * 16 + rowOff + i;
      int col = h * DV + g * 16 + nIdx;
      attn[(size_t)row * (NH * DV) + col] = f2bf(accO[g][i] / Lr[i]);
    }
  }
}

// ---------------------------------------------------------------------------
extern "C" void kernel_launch(void* const* d_in, const int* in_sizes, int n_in,
                              void* d_out, int out_size, void* d_ws, size_t ws_size,
                              hipStream_t stream) {
  (void)in_sizes; (void)n_in; (void)out_size; (void)ws_size;
  const float* x     = (const float*)d_in[0];
  const float* fcos  = (const float*)d_in[1];
  const float* fsin  = (const float*)d_in[2];
  const float* wq_a  = (const float*)d_in[3];
  const float* q_ln  = (const float*)d_in[4];
  const float* wq_b  = (const float*)d_in[5];
  const float* wkv_a = (const float*)d_in[6];
  const float* kv_ln = (const float*)d_in[7];
  const float* wkv_b = (const float*)d_in[8];
  const float* wo    = (const float*)d_in[9];
  float* out = (float*)d_out;

  char* ws = (char*)d_ws;
  size_t off = 0;
  auto take = [&](size_t bytes) -> char* {
    char* p = ws + off;
    off += (bytes + 255) & ~(size_t)255;
    return p;
  };
  unsigned short* x_bf    = (unsigned short*)take((size_t)S_LEN * DIM * 2);
  unsigned short* wqa_bf  = (unsigned short*)take((size_t)QLORA * DIM * 2);
  unsigned short* wqb_bf  = (unsigned short*)take((size_t)(NH * DQK) * QLORA * 2);
  unsigned short* wkva_bf = (unsigned short*)take((size_t)576 * DIM * 2);
  unsigned short* wkvb_bf = (unsigned short*)take((size_t)(NH * 256) * KVLORA * 2);
  unsigned short* wo_bf   = (unsigned short*)take((size_t)DIM * (NH * DV) * 2);
  unsigned short* qn_bf   = (unsigned short*)take((size_t)S_LEN * QLORA * 2);
  unsigned short* qk_bf   = (unsigned short*)take((size_t)S_LEN * NH * DQK * 2);
  unsigned short* kvc_bf  = (unsigned short*)take((size_t)S_LEN * KVLORA * 2);
  float*          kpe_f   = (float*)take((size_t)S_LEN * 64 * 4);
  unsigned short* k_bf    = (unsigned short*)take((size_t)S_LEN * NH * DQK * 2);
  unsigned short* vt_bf   = (unsigned short*)take((size_t)NH * DV * S_LEN * 2);
  unsigned short* attn_bf = (unsigned short*)take((size_t)S_LEN * NH * DV * 2);
  float*          scratch = (float*)take((size_t)S_LEN * 4096 * 4); // reused

  auto cvt = [&](const float* src, unsigned short* dst, int n) {
    mla_cvt_bf16<<<(n + 255) / 256, 256, 0, stream>>>(src, dst, n);
  };
  cvt(x,     x_bf,    S_LEN * DIM);
  cvt(wq_a,  wqa_bf,  QLORA * DIM);
  cvt(wq_b,  wqb_bf,  NH * DQK * QLORA);
  cvt(wkv_a, wkva_bf, 576 * DIM);
  cvt(wkv_b, wkvb_bf, NH * 256 * KVLORA);
  cvt(wo,    wo_bf,   DIM * NH * DV);

  // q path
  mla_gemm_nt<2><<<dim3(QLORA / 128, S_LEN / 64), 256, 0, stream>>>(
      x_bf, wqa_bf, scratch, S_LEN, QLORA, DIM);
  mla_rmsnorm<<<S_LEN, 256, 0, stream>>>(scratch, q_ln, qn_bf, QLORA);
  mla_gemm_nt<2><<<dim3((NH * DQK) / 128, S_LEN / 64), 256, 0, stream>>>(
      qn_bf, wqb_bf, scratch, S_LEN, NH * DQK, QLORA);
  mla_build_q<<<S_LEN, 256, 0, stream>>>(scratch, fcos, fsin, qk_bf);

  // kv path (N=576 -> 64-wide block tiles)
  mla_gemm_nt<1><<<dim3(576 / 64, S_LEN / 128), 256, 0, stream>>>(
      x_bf, wkva_bf, scratch, S_LEN, 576, DIM);
  mla_kv_norm_rope<<<S_LEN, 256, 0, stream>>>(scratch, kv_ln, fcos, fsin,
                                              kvc_bf, kpe_f);
  mla_gemm_nt<2><<<dim3((NH * 256) / 128, S_LEN / 64), 256, 0, stream>>>(
      kvc_bf, wkvb_bf, scratch, S_LEN, NH * 256, KVLORA);
  mla_build_k<<<S_LEN, 256, 0, stream>>>(scratch, kpe_f, k_bf, vt_bf);

  // attention
  mla_attn<<<dim3(S_LEN / 16, NH), 32, 0, stream>>>(qk_bf, k_bf, vt_bf, attn_bf);

  // output projection -> f32
  mla_gemm_nt<2><<<dim3(DIM / 128, S_LEN / 64), 256, 0, stream>>>(
      attn_bf, wo_bf, out, S_LEN, DIM, NH * DV);
}
